// MyParticleNetwork_61323543052521
// MI455X (gfx1250) — matrix operand order repaired and checked by reading, not compiled, not run on hardware
//
#include <hip/hip_runtime.h>

// ===========================================================================
// MyParticleNetwork forward on gfx1250 (MI455X).
// Matrix work (continuous-conv einsum + dense layers) runs on
// v_wmma_f32_16x16x32_f16 (f16 inputs, f32 accumulate).
// Weights are pre-packed into per-lane WMMA fragment order (one 32B vector
// load per lane per WMMA). Neighbor features are prefetched chunk-ahead with
// GLOBAL_LOAD_ASYNC_TO_LDS_B128 (ASYNCcnt double-buffer pipeline).
// ===========================================================================

#define NP   8192
#define MP   4096
#define KNB  16
#define DTS  0.02f
#define RRAD 0.1125f          // EXTENT * 0.5

typedef __attribute__((ext_vector_type(16))) _Float16 v16h;
typedef __attribute__((ext_vector_type(8)))  float    v8f;
typedef _Float16 half_t;

__device__ __forceinline__ float signf(float v){ return v>0.f?1.f:(v<0.f?-1.f:0.f); }

// 16B async global->LDS copy (per lane), tracked by ASYNCcnt
__device__ __forceinline__ void asyncB128(const float* gsrc, float* ldst){
  unsigned loff = (unsigned)(unsigned long long)ldst;   // low 32b of generic = LDS offset
  asm volatile("global_load_async_to_lds_b128 %0, %1, off"
               :: "v"(loff), "v"(gsrc) : "memory");
}
__device__ __forceinline__ void waitAsyncLE4(){
  asm volatile("s_wait_asynccnt 0x4" ::: "memory");
}
__device__ __forceinline__ void waitAsyncLE0(){
  asm volatile("s_wait_asynccnt 0x0" ::: "memory");
}

// ---------------------------------------------------------------------------
// integrate: vel2 = vel + dt*g ; pos2 = pos + dt*(vel2+vel)/2 ; feats=[1,vel2]
// ---------------------------------------------------------------------------
__global__ void k_integrate(const float* __restrict__ pos, const float* __restrict__ vel,
                            float* __restrict__ pos2, float* __restrict__ feats, int n){
  int i = blockIdx.x*blockDim.x + threadIdx.x;
  if (i >= n) return;
  float vx = vel[i*3+0], vy = vel[i*3+1], vz = vel[i*3+2];
  float wx = vx, wy = vy - 9.81f*DTS, wz = vz;
  pos2[i*3+0] = pos[i*3+0] + DTS*0.5f*(wx+vx);
  pos2[i*3+1] = pos[i*3+1] + DTS*0.5f*(wy+vy);
  pos2[i*3+2] = pos[i*3+2] + DTS*0.5f*(wz+vz);
  feats[i*4+0] = 1.f; feats[i*4+1] = wx; feats[i*4+2] = wy; feats[i*4+3] = wz;
}

// ---------------------------------------------------------------------------
// Pack weights into WMMA B-fragment consumption order (f32 -> f16).
// Output layout: halves[((ct*nkbT + s)*32 + lane)*16 + h]
//   col = ct*16 + (lane&15)          (zero-padded past Cout)
//   cconv: step s -> chunk ch = s/(2*CC), kb = (s%(2*CC))*32,
//          kbk = kb + (h<8?0:16) + 8*(lane>>4) + (h&7), l = kbk>>ccs,
//          c = (kbk & (CC-1)) + ch*CC, row = l*Cin + c (zero if c >= Cin)
//   dense: k = s*32 + (h<8?0:16) + 8*(lane>>4) + (h&7), row = k (zero if k >= Cin)
// ---------------------------------------------------------------------------
__global__ void k_packw(const float* __restrict__ w32, half_t* __restrict__ wp,
                        int Cin, int ccs, int Cout, int nkbT, int isDense){
  int idx = blockIdx.x*blockDim.x + threadIdx.x;
  int ntiles = (Cout + 15) >> 4;
  int total = ntiles*nkbT*32;
  if (idx >= total) return;
  int lane = idx & 31;
  int t = idx >> 5;
  int s = t % nkbT;
  int hi = lane >> 4, lm = lane & 15;
  int ct = t / nkbT;
  int col = ct*16 + lm;
  int CC = 1 << ccs;
  int spc = 2*CC;                       // wmma steps per Cin-chunk
  v16h v;
  #pragma unroll
  for (int h = 0; h < 16; ++h){
    float val = 0.f;
    if (col < Cout){
      if (isDense){
        int k = s*32 + ((h<8)?0:16) + 8*hi + (h&7);
        if (k < Cin) val = w32[(size_t)k*Cout + col];
      } else {
        int ch = s / spc;
        int kb = (s % spc)*32;
        int kbk = kb + ((h<8)?0:16) + 8*hi + (h&7);
        int l = kbk >> ccs;
        int c = (kbk & (CC-1)) + (ch << ccs);
        if (c < Cin) val = w32[((size_t)l*Cin + c)*Cout + col];
      }
    }
    v[h] = (half_t)val;
  }
  *(v16h*)(wp + (size_t)idx*16) = v;
}

// ---------------------------------------------------------------------------
// Fused continuous convolution (16 query rows / block, 256 threads = 8 waves)
// ---------------------------------------------------------------------------
#define CCMAX 16
#define ASTMAX (64*CCMAX + 4)     // padded row stride (floats); 16B aligned

__global__ void __launch_bounds__(256)
k_cconv(const float* __restrict__ feat, int Cin, int Cpad, int ccs,
        const float* __restrict__ srcpos, const float* __restrict__ qpos,
        const int* __restrict__ nbrs, const half_t* __restrict__ Wp,
        const float* __restrict__ bias, float* __restrict__ out, int Cout,
        int ignoreSelf)
{
  __shared__ __align__(16) float sA[16*ASTMAX];      // A tile [row][l*CC + c]
  __shared__ __align__(16) float sF[2*256*CCMAX];    // double-buffered features
  __shared__ float        sW8[256*8];                // window*trilinear weights
  __shared__ unsigned char sC8[256*8];               // cell index per corner
  __shared__ int          sNb[256];

  const int tid = threadIdx.x;
  const int n0  = blockIdx.x*16;
  const int CC  = 1 << ccs;                  // 4 or 16
  const int chunks = Cpad >> ccs;
  const int AST = 64*CC + 4;
  const bool useAsync = (CC == 16);          // Cin multiple of 16 -> 16B aligned rows

  // ---- Phase A: geometry -------------------------------------------------
  {
    int r = tid >> 4, kk = tid & 15;
    int n = n0 + r;
    int nb = nbrs[n*KNB + kk];
    sNb[tid] = nb;
    float dx = srcpos[nb*3+0] - qpos[n*3+0];
    float dy = srcpos[nb*3+1] - qpos[n*3+1];
    float dz = srcpos[nb*3+2] - qpos[n*3+2];
    const float inv = 1.f / RRAD;
    float px = dx*inv, py = dy*inv, pz = dz*inv;
    float r2 = px*px + py*py + pz*pz;
    float w = 1.f - r2; w = w*w*w;
    w = fminf(fmaxf(w, 0.f), 1.f);
    bool valid = (r2 < 1.f);
    if (ignoreSelf && nb == n) valid = false;
    if (!valid) w = 0.f;
    // ball -> cube (volume preserving)
    const float eps = 1e-9f;
    float nn  = sqrtf(px*px + py*py + pz*pz + eps);
    float xy2 = px*px + py*py;
    bool  c1  = (1.25f*pz*pz) > xy2;
    float s1  = sqrtf(3.f*nn / (nn + fabsf(pz) + eps));
    float s2  = nn * rsqrtf(xy2 + eps);
    float cx  = c1 ? px*s1 : px*s2;
    float cy  = c1 ? py*s1 : py*s2;
    float cz  = c1 ? signf(pz)*nn : 1.5f*pz;
    float rxy = sqrtf(cx*cx + cy*cy + eps);
    bool  c2  = fabsf(cy) <= fabsf(cx);
    const float k4 = 4.f / 3.14159265358979323846f;
    float ax = fabsf(cx) + eps, ay = fabsf(cy) + eps;
    float ux = c2 ? signf(cx)*rxy : k4*rxy*atanf(cx/ay);
    float uy = c2 ? k4*rxy*atanf(cy/ax) : signf(cy)*rxy;
    float u0 = fminf(fmaxf(ux, -1.f), 1.f);
    float u1 = fminf(fmaxf(uy, -1.f), 1.f);
    float u2 = fminf(fmaxf(cz, -1.f), 1.f);
    float g0 = (u0+1.f)*1.5f, g1 = (u1+1.f)*1.5f, g2 = (u2+1.f)*1.5f;
    int i0x = min(max((int)floorf(g0),0),3);
    int i0y = min(max((int)floorf(g1),0),3);
    int i0z = min(max((int)floorf(g2),0),3);
    int i1x = min(i0x+1,3), i1y = min(i0y+1,3), i1z = min(i0z+1,3);
    float fx = g0-(float)i0x, fy = g1-(float)i0y, fz = g2-(float)i0z;
    #pragma unroll
    for (int c8 = 0; c8 < 8; ++c8){
      int bx = c8&1, by = (c8>>1)&1, bz = c8>>2;
      int ix = bx?i1x:i0x, iy = by?i1y:i0y, iz = bz?i1z:i0z;
      float wx = bx?fx:1.f-fx, wy = by?fy:1.f-fy, wz = bz?fz:1.f-fz;
      sC8[tid*8+c8] = (unsigned char)((iz*4+iy)*4+ix);
      sW8[tid*8+c8] = w*wx*wy*wz;
    }
  }

  const int wid = tid >> 5;      // wave id (0..7) = 16-col tile of Cout
  const int lane = tid & 31;
  const int hi = lane >> 4;
  const int lm = lane & 15;
  const int colBase = wid*16;
  const bool participate = colBase < Cout;
  const int nkbT = 2*Cpad;                       // total wmma steps
  const half_t* wpTile = Wp + ((size_t)wid*nkbT*32 + lane)*16;
  v8f acc = {0.f,0.f,0.f,0.f,0.f,0.f,0.f,0.f};

  // prefetch chunk 0 features (async, in-order completion, per-wave visible)
  const int nbMine = sNb[tid];
  if (useAsync){
    const float* g = feat + (size_t)nbMine*Cin;    // chunk 0: cb = 0
    float* l = sF + tid*CCMAX;                     // buffer 0
    asyncB128(g + 0, l + 0);
    asyncB128(g + 4, l + 4);
    asyncB128(g + 8, l + 8);
    asyncB128(g + 12, l + 12);
  }

  // ---- Phase B: chunk loop ----------------------------------------------
  for (int ch = 0; ch < chunks; ++ch){
    __syncthreads();                 // prev GEMM done: sA free, old sF buf free
    // zero A tile (incl. padding)
    for (int i = tid; i < 16*AST; i += 256) sA[i] = 0.f;
    const int buf = useAsync ? (ch & 1) : 0;
    if (useAsync){
      bool more = (ch + 1) < chunks;
      if (more){                     // prefetch next chunk into other buffer
        const float* g = feat + (size_t)nbMine*Cin + ((ch+1) << ccs);
        float* l = sF + ((ch+1)&1)*256*CCMAX + tid*CCMAX;
        asyncB128(g + 0, l + 0);
        asyncB128(g + 4, l + 4);
        asyncB128(g + 8, l + 8);
        asyncB128(g + 12, l + 12);
        waitAsyncLE4();              // oldest 4 (current chunk) complete
      } else {
        waitAsyncLE0();
      }
    } else {
      // synchronous staging (tiny Cin: 3 or 4, single chunk)
      for (int c = 0; c < CC; ++c)
        sF[tid*CCMAX + c] = (c < Cin) ? feat[(size_t)nbMine*Cin + c] : 0.f;
    }
    __syncthreads();
    // scatter: 16 threads per row, channel = t; serialized over k (no atomics)
    {
      int r = tid >> 4, t = tid & 15;
      if (t < CC){
        float* Arow = sA + r*AST;
        const float* Fb = sF + buf*256*CCMAX;
        for (int kk = 0; kk < KNB; ++kk){
          int pair = r*16 + kk;
          float fj = Fb[pair*CCMAX + t];
          #pragma unroll
          for (int c8 = 0; c8 < 8; ++c8){
            float wv = sW8[pair*8+c8];
            int cell  = sC8[pair*8+c8];
            Arow[cell*CC + t] += fj*wv;
          }
        }
      }
    }
    __syncthreads();
    // WMMA: A from LDS (4x ds_load_b128 + pk-cvt), B one 32B load per lane
    if (participate){
      const int spc = 2*CC;
      const int sBase = ch*spc;
      const float* Arow = sA + lm*AST + (hi<<3);
      for (int sl = 0; sl < spc; ++sl){
        v16h b = *(const v16h*)(wpTile + (size_t)(sBase + sl)*512);
        const float* A0 = Arow + sl*32;
        float4 p0 = *(const float4*)(A0);
        float4 p1 = *(const float4*)(A0 + 4);
        float4 p2 = *(const float4*)(A0 + 16);
        float4 p3 = *(const float4*)(A0 + 20);
        v16h a;
        a[0]=(half_t)p0.x;  a[1]=(half_t)p0.y;  a[2]=(half_t)p0.z;  a[3]=(half_t)p0.w;
        a[4]=(half_t)p1.x;  a[5]=(half_t)p1.y;  a[6]=(half_t)p1.z;  a[7]=(half_t)p1.w;
        a[8]=(half_t)p2.x;  a[9]=(half_t)p2.y;  a[10]=(half_t)p2.z; a[11]=(half_t)p2.w;
        a[12]=(half_t)p3.x; a[13]=(half_t)p3.y; a[14]=(half_t)p3.z; a[15]=(half_t)p3.w;
        acc = __builtin_amdgcn_wmma_f32_16x16x32_f16(false, a, false, b,
                                                     (short)0, acc, false, false);
      }
    }
  }

  // ---- store: C/D layout: lane col=lm, rows m = rr + 8*hi ------------------
  if (participate){
    int col = colBase + lm;
    if (col < Cout){
      float bv = bias[col];
      #pragma unroll
      for (int rr = 0; rr < 8; ++rr){
        int m = rr + (hi<<3);
        out[(size_t)(n0+m)*Cout + col] = acc[rr] + bv;
      }
    }
  }
}

// ---------------------------------------------------------------------------
// Dense GEMM: out[N,Cout] (+)= inp[N,Cin] * W[Cin,Cout] + b  (1 wave / tile)
// ---------------------------------------------------------------------------
__global__ void __launch_bounds__(32)
k_dense(const float* __restrict__ inp, int Cin,
        const half_t* __restrict__ Wp, const float* __restrict__ bias,
        float* __restrict__ out, int Cout, int accumulate)
{
  int lane = threadIdx.x & 31;
  int n0 = blockIdx.x*16;
  int hi = lane >> 4, lm = lane & 15;
  int col = blockIdx.y*16 + lm;
  bool colOK = col < Cout;
  int kPad = (Cin + 31) & ~31;
  int nkbT = kPad >> 5;
  const half_t* wpTile = Wp + ((size_t)blockIdx.y*nkbT*32 + lane)*16;
  const float* irow = inp + (size_t)(n0+lm)*Cin + (hi<<3);
  v8f acc = {0.f,0.f,0.f,0.f,0.f,0.f,0.f,0.f};
  for (int s = 0; s < nkbT; ++s){
    v16h b = *(const v16h*)(wpTile + (size_t)s*512);
    int kb = s*32;
    v16h a;
    if (kb + 32 <= Cin){
      const float* A0 = irow + kb;
      float4 p0 = *(const float4*)(A0);
      float4 p1 = *(const float4*)(A0 + 4);
      float4 p2 = *(const float4*)(A0 + 16);
      float4 p3 = *(const float4*)(A0 + 20);
      a[0]=(half_t)p0.x;  a[1]=(half_t)p0.y;  a[2]=(half_t)p0.z;  a[3]=(half_t)p0.w;
      a[4]=(half_t)p1.x;  a[5]=(half_t)p1.y;  a[6]=(half_t)p1.z;  a[7]=(half_t)p1.w;
      a[8]=(half_t)p2.x;  a[9]=(half_t)p2.y;  a[10]=(half_t)p2.z; a[11]=(half_t)p2.w;
      a[12]=(half_t)p3.x; a[13]=(half_t)p3.y; a[14]=(half_t)p3.z; a[15]=(half_t)p3.w;
    } else {
      #pragma unroll
      for (int h = 0; h < 16; ++h){
        int ka = kb + ((h<8)?0:16) + (hi<<3) + (h&7);
        a[h] = (ka < Cin) ? (half_t)inp[(size_t)(n0+lm)*Cin + ka] : (half_t)0.f;
      }
    }
    acc = __builtin_amdgcn_wmma_f32_16x16x32_f16(false, a, false, b,
                                                 (short)0, acc, false, false);
  }
  if (colOK){
    float bv = bias[col];
    #pragma unroll
    for (int rr = 0; rr < 8; ++rr){
      int m = rr + (hi<<3);
      size_t o = (size_t)(n0+m)*Cout + col;
      float v = acc[rr] + bv;
      out[o] = accumulate ? (out[o] + v) : v;
    }
  }
}

// ---------------------------------------------------------------------------
// BatchNorm (training-mode batch stats) + activation, elementwise glue
// ---------------------------------------------------------------------------
__global__ void __launch_bounds__(256)
k_bnstats(const float* __restrict__ x, int n, int C,
          float* __restrict__ mean, float* __restrict__ rstd){
  __shared__ float ss[256], sq[256];
  int c = blockIdx.x;
  float s = 0.f, q = 0.f;
  for (int i = threadIdx.x; i < n; i += 256){
    float v = x[(size_t)i*C + c]; s += v; q += v*v;
  }
  ss[threadIdx.x] = s; sq[threadIdx.x] = q;
  __syncthreads();
  for (int st = 128; st > 0; st >>= 1){
    if (threadIdx.x < st){ ss[threadIdx.x] += ss[threadIdx.x+st];
                           sq[threadIdx.x] += sq[threadIdx.x+st]; }
    __syncthreads();
  }
  if (threadIdx.x == 0){
    float m = ss[0]/(float)n;
    float var = sq[0]/(float)n - m*m;
    mean[c] = m; rstd[c] = rsqrtf(var + 1e-5f);
  }
}

__global__ void k_bnact(const float* __restrict__ x, float* __restrict__ y, int n, int C,
                        const float* __restrict__ mean, const float* __restrict__ rstd,
                        const float* __restrict__ g, const float* __restrict__ b, int act){
  int idx = blockIdx.x*blockDim.x + threadIdx.x;
  if (idx >= n*C) return;
  int c = idx % C;
  float v = (x[idx] - mean[c])*rstd[c]*g[c] + b[c];
  if (act == 1) v = fmaxf(v, 0.f);
  else if (act == 2) v = 1.f/(1.f + expf(-v));
  y[idx] = v;
}

__global__ void k_relu(const float* __restrict__ x, float* __restrict__ y, int n){
  int i = blockIdx.x*blockDim.x + threadIdx.x;
  if (i < n) y[i] = fmaxf(x[i], 0.f);
}

__global__ void k_concat(const float* __restrict__ x, int cx,
                         const float* __restrict__ y, int cy,
                         float* __restrict__ out, int n){
  int idx = blockIdx.x*blockDim.x + threadIdx.x;
  int C = cx + cy;
  if (idx >= n*C) return;
  int i = idx / C, c = idx % C;
  out[idx] = (c < cx) ? x[(size_t)i*cx + c] : y[(size_t)i*cy + (c - cx)];
}

__global__ void k_combine(const float* __restrict__ x, const float* __restrict__ y,
                          const float* __restrict__ w, float* __restrict__ out, int n){
  int i = blockIdx.x*blockDim.x + threadIdx.x;
  if (i >= n) return;
  float wv = w[i];
  out[i] = 2.f*x[i]*wv + 2.f*y[i]*(1.f - wv);
}

__global__ void k_final(const float* __restrict__ pos, const float* __restrict__ pos2,
                        const float* __restrict__ ans, float* __restrict__ out, int n){
  int i = blockIdx.x*blockDim.x + threadIdx.x;
  if (i >= n) return;
  #pragma unroll
  for (int j = 0; j < 3; ++j){
    float pn = pos2[i*3+j] + ans[i*3+j]*(1.f/128.f);
    out[i*6+j]   = pn;
    out[i*6+3+j] = (pn - pos[i*3+j])*(1.f/DTS);
  }
}

// ===========================================================================
// Host orchestration
// ===========================================================================
namespace {
struct CvP { const float* w; const float* b; int cin; int cout; };
struct BnPr{ const float* g; const float* b; };
struct DnP { const float* w; const float* b; int cin; int cout; };
struct AffPr { CvP c1, c2; BnPr bn1, bn2; int ch, inter; };
struct BrP { CvP c0f, c0o; CvP cv[4]; DnP d0; DnP dn[4];
             CvP ic1, ic2, ic3, ic4; BnPr ib1, ib2, ib3, ib4; };
inline int G256(long n){ return (int)((n + 255)/256); }
}

extern "C" void kernel_launch(void* const* d_in, const int* in_sizes, int n_in,
                              void* d_out, int out_size, void* d_ws, size_t ws_size,
                              hipStream_t stream) {
  (void)in_sizes; (void)n_in; (void)out_size; (void)ws_size;
  const int N = NP;

  // ---- input indexing ------------------------------------------------------
  // jax tree flattening of params: sorted dict keys, lists in order,
  // leaf order within conv={b,w}, bn={b,g}, dense={b,w}.
  // top-level: params tree (124 leaves), pos, vel, box, box_feats, nbr_ff, nbr_fo
  auto F = [&](int i){ return (const float*)d_in[i]; };
  auto I = [&](int i){ return (const int*)d_in[i]; };
  const int chans[4] = {64, 128, 64, 3};
  const int cins[4]  = {64, 64, 128, 64};

  auto getAff = [&](int base, int ch, int inter){
    AffPr a;
    a.bn1 = { F(base+1), F(base+0) };
    a.bn2 = { F(base+3), F(base+2) };
    a.c1  = { F(base+5), F(base+4), 2*ch, inter };
    a.c2  = { F(base+7), F(base+6), inter, ch };
    a.ch = ch; a.inter = inter; return a;
  };
  auto getBr = [&](int b){
    BrP p;
    p.c0f = { F(b+1), F(b+0), 4, 32 };
    p.c0o = { F(b+3), F(b+2), 3, 32 };
    for (int j = 0; j < 4; ++j) p.cv[j] = { F(b+5+2*j), F(b+4+2*j), cins[j], chans[j] };
    p.d0 = { F(b+13), F(b+12), 4, 32 };
    for (int j = 0; j < 4; ++j) p.dn[j] = { F(b+15+2*j), F(b+14+2*j), cins[j], chans[j] };
    int ib = b + 22;
    p.ib1 = { F(ib+1),  F(ib+0) };  p.ib2 = { F(ib+3),  F(ib+2) };
    p.ib3 = { F(ib+5),  F(ib+4) };  p.ib4 = { F(ib+7),  F(ib+6) };
    p.ic1 = { F(ib+9),  F(ib+8),  64, 64 };
    p.ic2 = { F(ib+11), F(ib+10), 64, 32 };
    p.ic3 = { F(ib+13), F(ib+12), 32, 64 };
    p.ic4 = { F(ib+15), F(ib+14), 64, 32 };
    return p;
  };

  AffPr aff0  = getAff(0, 64, 64);
  AffPr affsP[4];
  for (int i = 0; i < 4; ++i) affsP[i] = getAff(8 + 8*i, chans[i], chans[i]);
  BrP brA = getBr(40);   // 'ascc'
  BrP brC = getBr(78);   // 'cconv'
  AffPr resAff = getAff(116, 64, 64);
  const float* pos  = F(124); const float* vel  = F(125);
  const float* box  = F(126); const float* boxf = F(127);
  const int* nbrff  = I(128); const int* nbrfo  = I(129);

  // ---- workspace layout ----------------------------------------------------
  float* ws = (float*)d_ws;
  size_t off = 0;
  auto alloc = [&](size_t n){ float* p = ws + off; off += n; return p; };
  float* pos2b = alloc((size_t)3*N);
  float* featb = alloc((size_t)4*N);
  float* braX  = alloc((size_t)64*N);
  float* braY  = alloc((size_t)64*N);
  float* t1 = alloc((size_t)256*N);
  float* t2 = alloc((size_t)128*N);
  float* t3 = alloc((size_t)128*N);
  float* t4 = alloc((size_t)128*N);
  float* t5 = alloc((size_t)128*N);
  float* t6 = alloc((size_t)128*N);
  float* s1 = alloc((size_t)128*N);
  float* s2 = alloc((size_t)128*N);
  float* s3 = alloc((size_t)128*N);
  float* s4 = alloc((size_t)128*N);
  float* meanb = alloc(256);
  float* rstdb = alloc(256);
  half_t* w16  = (half_t*)alloc((size_t)1100*1024);   // packed cconv W (<=4MB)
  half_t* w16d = (half_t*)alloc((size_t)  80*1024);   // packed dense W

  // ---- composite-op helpers ------------------------------------------------
  auto cconv = [&](const float* feat, int Cin, const float* sp, const float* qp,
                   const int* nb, const CvP& cv, float* outb, bool selfmask){
    int Cpad = (Cin == 3) ? 4 : Cin;
    int ccs  = (Cpad == 4) ? 2 : 4;            // CC = 4 or 16
    int nkbT = 2*Cpad;
    int ntiles = (cv.cout + 15)/16;
    long tot = (long)ntiles*nkbT*32;
    k_packw<<<G256(tot), 256, 0, stream>>>(cv.w, w16, Cin, ccs, cv.cout, nkbT, 0);
    k_cconv<<<N/16, 256, 0, stream>>>(feat, Cin, Cpad, ccs, sp, qp, nb, w16,
                                      cv.b, outb, cv.cout, selfmask ? 1 : 0);
  };
  auto dense = [&](const float* inp, int Cin, const DnP& dp, float* outb, bool acc){
    int kPad = (Cin + 31) & ~31;
    int nkbT = kPad >> 5;
    int ntiles = (dp.cout + 15)/16;
    long tot = (long)ntiles*nkbT*32;
    k_packw<<<G256(tot), 256, 0, stream>>>(dp.w, w16d, Cin, 4, dp.cout, nkbT, 1);
    dim3 grid(N/16, ntiles);
    k_dense<<<grid, 32, 0, stream>>>(inp, Cin, w16d, dp.b, outb, dp.cout, acc ? 1 : 0);
  };
  auto bnact = [&](float* x, int C, const BnPr& bn, int act){
    k_bnstats<<<C, 256, 0, stream>>>(x, N, C, meanb, rstdb);
    k_bnact<<<G256((long)N*C), 256, 0, stream>>>(x, x, N, C, meanb, rstdb, bn.g, bn.b, act);
  };
  auto concat = [&](const float* x, int cx, const float* y, int cy, float* outb){
    k_concat<<<G256((long)N*(cx+cy)), 256, 0, stream>>>(x, cx, y, cy, outb, N);
  };
  auto combine = [&](const float* x, const float* y, const float* w, int C, float* outb){
    k_combine<<<G256((long)N*C), 256, 0, stream>>>(x, y, w, outb, N*C);
  };
  auto aff = [&](const AffPr& a, const float* x, const float* y, float* outb){
    concat(x, a.ch, y, a.ch, t1);
    cconv(t1, 2*a.ch, pos2b, pos2b, nbrff, a.c1, s1, true);
    bnact(s1, a.inter, a.bn1, 1 /*relu*/);
    cconv(s1, a.inter, pos2b, pos2b, nbrff, a.c2, s2, true);
    bnact(s2, a.ch, a.bn2, 2 /*sigmoid*/);
    combine(x, y, s2, a.ch, outb);
  };
  auto runBranch = [&](const BrP& bp, float* outb){
    cconv(featb, 4, pos2b, pos2b, nbrff, bp.c0f, s1, true);     // c0 [N,32]
    cconv(boxf, 3, box,   pos2b, nbrfo, bp.c0o, s2, false);     // ob [N,32]
    dense(featb, 4, bp.d0, s3, false);                          // d0 [N,32]
    // iAFF(x=c0, y=ob), ch=32, inter=64
    concat(s1, 32, s2, 32, t1);
    cconv(t1, 64, pos2b, pos2b, nbrff, bp.ic1, t2, true); bnact(t2, 64, bp.ib1, 1);
    cconv(t2, 64, pos2b, pos2b, nbrff, bp.ic2, t3, true); bnact(t3, 32, bp.ib2, 2);
    combine(s1, s2, t3, 32, s4);                                // xo
    cconv(s4, 32, pos2b, pos2b, nbrff, bp.ic3, t2, true); bnact(t2, 64, bp.ib3, 1);
    cconv(t2, 64, pos2b, pos2b, nbrff, bp.ic4, t3, true); bnact(t3, 32, bp.ib4, 2);
    combine(s1, s2, t3, 32, s4);                                // fused
    concat(s4, 32, s3, 32, outb);                               // [N,64]
  };

  // ---- forward graph -------------------------------------------------------
  k_integrate<<<G256(N), 256, 0, stream>>>(pos, vel, pos2b, featb, N);
  runBranch(brC, braX);
  runBranch(brA, braY);
  aff(aff0, braX, braY, t4);                                    // ans [N,64]
  for (int i = 0; i < 4; ++i){
    const float* inp;
    int cin = cins[i];
    if (i == 0) inp = t4;
    else { k_relu<<<G256((long)N*cin), 256, 0, stream>>>(t4, t6, N*cin); inp = t6; }
    cconv(inp, cin, pos2b, pos2b, nbrff, brC.cv[i], t2, true);
    dense(inp, cin, brC.dn[i], t2, true);
    cconv(inp, cin, pos2b, pos2b, nbrff, brA.cv[i], t3, true);
    dense(inp, cin, brA.dn[i], t3, true);
    aff(affsP[i], t2, t3, t4);                                  // ans [N,chans[i]]
    if (i == 0)
      hipMemcpyAsync(t5, t4, (size_t)N*64*sizeof(float), hipMemcpyDeviceToDevice, stream);
    if (i == 2)
      aff(resAff, t4, t5, t4);
  }
  k_final<<<G256(N), 256, 0, stream>>>(pos, pos2b, t4, (float*)d_out, N);
}
